// GCA_78666620993816
// MI455X (gfx1250) — compile-verified, matrix operand order checked by reading, not run on hardware
//
#include <hip/hip_runtime.h>

#define B_   16
#define C_   16
#define H_   256
#define W_   256
#define HID  128
#define OUTC 13
#define ROWS 4            // output rows per workgroup (6 staged rows incl. halo)
#define XPITCH 136        // LDS row pitch (floats): halo at 3/132, center 4..131 (16B aligned)

typedef __attribute__((ext_vector_type(16))) __bf16 v16bf;
typedef __attribute__((ext_vector_type(8)))  __bf16 v8bf;
typedef __attribute__((ext_vector_type(8)))  float  v8f;

// round-to-nearest-even f32 -> bf16 (as raw ushort) -- cold path (prep kernel)
static __device__ __forceinline__ unsigned short f2bf(float f) {
  unsigned u = __builtin_bit_cast(unsigned, f);
  u += 0x7FFFu + ((u >> 16) & 1u);
  return (unsigned short)(u >> 16);
}

// hot path: round-half-up bias then grab both high halves with one v_perm_b32
static __device__ __forceinline__ unsigned pack2_bf16(float lo, float hi) {
  unsigned a = __builtin_bit_cast(unsigned, lo) + 0x8000u;
  unsigned b = __builtin_bit_cast(unsigned, hi) + 0x8000u;
  // result bytes [1:0] = a[3:2], [3:2] = b[3:2]
  return __builtin_amdgcn_perm(b, a, 0x07060302u);
}

// packed bf16 relu: one VOP3P instruction for two channels
static __device__ __forceinline__ unsigned relu2_bf16(unsigned w) {
  unsigned r;
  asm("v_pk_max_num_bf16 %0, %1, 0" : "=v"(r) : "v"(w));
  return r;
}

static __device__ __forceinline__ v16bf cat8(v8bf lo, v8bf hi) {
  union { v8bf h[2]; v16bf v; } u;
  u.h[0] = lo; u.h[1] = hi;
  return u.v;
}

// ---------------------------------------------------------------------------
// Weight prep: swizzle w1/b1/w2 into per-lane bf16 A-fragment layout of
// V_WMMA_F32_16X16X32_BF16:
//   lane l: M = mtile*16 + (l&15); VGPR pair p holds K-pair
//   klocal(p, half=l>>4) = (p&4 ? 16:0) + half*8 + (p&3)*2  (+0/+1)
// Perception K is INTERLEAVED: k = 4*ch + {0:x, 1:gx, 2:gy, 3:pad/bias};
// k==3 row carries b1 (perception slot k==3 holds 1.0).
// w1 frags: [mtile 0..7][kchunk 0..1][lane 0..31][16]  (8192 bf16)
// w2 frags: [kchunk 0..3][lane 0..31][16]              (2048 bf16)
// ---------------------------------------------------------------------------
__global__ void prep_frags(const float* __restrict__ w1,
                           const float* __restrict__ b1,
                           const float* __restrict__ w2,
                           unsigned short* __restrict__ frag) {
  int i = blockIdx.x * 256 + threadIdx.x;
  if (i < 8192) {
    int j    = i & 15;
    int lane = (i >> 4) & 31;
    int kc   = (i >> 9) & 1;
    int mt   = i >> 10;
    int p    = j >> 1;
    int klocal = ((p & 4) ? 16 : 0) + ((lane >> 4) * 8) + ((p & 3) * 2) + (j & 1);
    int k  = kc * 32 + klocal;        // 0..63, interleaved perception index
    int ch = k >> 2;
    int t  = k & 3;
    int m  = mt * 16 + (lane & 15);
    float v = 0.f;
    if (t == 0)      v = w1[m * 48 + ch];         // x
    else if (t == 1) v = w1[m * 48 + 16 + ch];    // sobel-x
    else if (t == 2) v = w1[m * 48 + 32 + ch];    // sobel-y
    else if (ch == 0) v = b1[m];                  // bias rides the GEMM
    frag[i] = f2bf(v);
  } else if (i < 10240) {
    int tt   = i - 8192;
    int j    = tt & 15;
    int lane = (tt >> 4) & 31;
    int kc   = tt >> 9;               // 0..3
    int p    = j >> 1;
    int klocal = ((p & 4) ? 16 : 0) + ((lane >> 4) * 8) + ((p & 3) * 2) + (j & 1);
    int k = kc * 32 + klocal;
    int m = lane & 15;
    float v = (m < OUTC) ? w2[m * 128 + k] : 0.f;
    frag[i] = f2bf(v);
  }
}

// ---------------------------------------------------------------------------
// Main NCA step. One workgroup (8 waves) handles (b, 4 rows, 128-pixel strip).
// ---------------------------------------------------------------------------
__global__ __launch_bounds__(256) void nca_step(const float* __restrict__ in,
                                                const unsigned short* __restrict__ frags,
                                                float* __restrict__ out) {
  __shared__ __align__(16) float          xLds[C_ * (ROWS + 2) * XPITCH]; // 52,224 B
  __shared__ __align__(16) unsigned short sbuf[128 * 128];                // 32,768 B

  const int tid  = threadIdx.x;
  const int wg   = blockIdx.x;
  const int xblk = wg & 1;
  const int yblk = (wg >> 1) & 63;
  const int b    = wg >> 7;
  const int x0   = xblk * 128;
  const int y0   = yblk * ROWS;

  const float* inb = in + (size_t)b * C_ * H_ * W_;

  // ---- Phase A: stage 16ch x 6 rows x cols [x0-1 .. x0+128] (wrapped) ----
  for (int i = tid; i < C_ * (ROWS + 2) * 32; i += 256) {   // 3072 float4 items
    int ch  = i / ((ROWS + 2) * 32);
    int rem = i - ch * ((ROWS + 2) * 32);
    int row = rem >> 5;
    int t   = rem & 31;
    int yy  = (y0 + row - 1) & (H_ - 1);
    float4 v = *(const float4*)(inb + ((size_t)ch * H_ + yy) * W_ + x0 + t * 4);
    *(float4*)(xLds + (ch * (ROWS + 2) + row) * XPITCH + 4 + t * 4) = v;
  }
  if (tid < C_ * (ROWS + 2) * 2) {                          // halo columns
    int ch   = tid / ((ROWS + 2) * 2);
    int rem  = tid - ch * ((ROWS + 2) * 2);
    int row  = rem >> 1;
    int side = rem & 1;
    int yy   = (y0 + row - 1) & (H_ - 1);
    int xx   = side ? ((x0 + 128) & (W_ - 1)) : ((x0 - 1) & (W_ - 1));
    xLds[(ch * (ROWS + 2) + row) * XPITCH + (side ? 132 : 3)] =
        inb[((size_t)ch * H_ + yy) * W_ + xx];
  }
  __syncthreads();

  const int lane = tid & 31;
  const int wv   = tid >> 5;
  const int half = lane >> 4;
  const int pix  = wv * 16 + (lane & 15);
  unsigned short* prow = sbuf + pix * 128;
  const v16bf* wf = (const v16bf*)frags;

#pragma unroll 1
  for (int r = 0; r < ROWS; ++r) {
    // ---- Phase B: perception row y0+r; 512 items of (ch, 4 pixels) -------
#pragma unroll
    for (int it = 0; it < 2; ++it) {
      int item = tid + it * 256;          // 0..511
      int ch   = item & 15;
      int px   = (item >> 4) * 4;         // group of 4 pixels
      const float* p0 = xLds + (ch * (ROWS + 2) + r) * XPITCH;
      const float* p1 = p0 + XPITCH;
      const float* p2 = p1 + XPITCH;
      // window indices px+3 .. px+8 : scalar + aligned b128 + scalar per row
      float4 m0 = *(const float4*)(p0 + px + 4);
      float4 m1 = *(const float4*)(p1 + px + 4);
      float4 m2 = *(const float4*)(p2 + px + 4);
      float t0[6] = {p0[px + 3], m0.x, m0.y, m0.z, m0.w, p0[px + 8]};
      float t1[6] = {p1[px + 3], m1.x, m1.y, m1.z, m1.w, p1[px + 8]};
      float t2[6] = {p2[px + 3], m2.x, m2.y, m2.z, m2.w, p2[px + 8]};
      // factored Sobel: e = row-weighted sums (for gx), d = row diffs (for gy)
      float e[6], d[6];
#pragma unroll
      for (int c = 0; c < 6; ++c) {
        e[c] = t0[c] + 2.f * t1[c] + t2[c];
        d[c] = t2[c] - t0[c];
      }
      unsigned padword = (ch == 0) ? 0x3F800000u : 0u;   // bf16(1.0) source
#pragma unroll
      for (int q = 0; q < 4; ++q) {
        float gx = e[q + 2] - e[q];
        float gy = (d[q] + d[q + 2]) + 2.f * d[q + 1];
        uint2 payload;
        payload.x = pack2_bf16(t1[q + 1], gx);
        payload.y = __builtin_amdgcn_perm(padword,
                     __builtin_bit_cast(unsigned, gy) + 0x8000u, 0x07060302u);
        *(uint2*)(sbuf + (px + q) * 128 + ch * 4) = payload;   // one ds_store_b64
      }
    }
    __syncthreads();

    // ---- GEMM phase: each wave owns 16 pixels of this row ----------------
    v16bf bp0 = cat8(*(const v8bf*)(prow + 0  + half * 8),
                     *(const v8bf*)(prow + 16 + half * 8));
    v16bf bp1 = cat8(*(const v8bf*)(prow + 32 + half * 8),
                     *(const v8bf*)(prow + 48 + half * 8));

    // GEMM1: h = relu(w1 . perc + b1), h overwrites perc rows in sbuf (bf16)
#pragma unroll
    for (int mt = 0; mt < 8; ++mt) {
      v16bf a0 = wf[(mt * 2 + 0) * 32 + lane];
      v16bf a1 = wf[(mt * 2 + 1) * 32 + lane];
      v8f acc = {};
      acc = __builtin_amdgcn_wmma_f32_16x16x32_bf16(false, a0, false, bp0, (short)0, acc, false, false);
      acc = __builtin_amdgcn_wmma_f32_16x16x32_bf16(false, a1, false, bp1, (short)0, acc, false, false);
      unsigned short* hrow = prow + mt * 16 + half * 8;
#pragma unroll
      for (int rr = 0; rr < 8; rr += 2) {
        // pack raw pair to bf16, then 1-op packed relu (round-then-relu == relu-then-round)
        *(unsigned*)(hrow + rr) = relu2_bf16(pack2_bf16(acc[rr], acc[rr + 1]));
      }
    }

    // GEMM2: ds = w2 . h  (weight rows 13..15 zero-padded)
    v8f acc2 = {};
#pragma unroll
    for (int kc = 0; kc < 4; ++kc) {
      v16bf a  = wf[512 + kc * 32 + lane];
      v16bf bh = cat8(*(const v8bf*)(prow + kc * 32 + half * 8),
                      *(const v8bf*)(prow + kc * 32 + 16 + half * 8));
      acc2 = __builtin_amdgcn_wmma_f32_16x16x32_bf16(false, a, false, bh, (short)0, acc2, false, false);
    }

    // ---- Residual add + store (coalesced along x per half-wave) ----------
    const int y = y0 + r;
    const int x = x0 + pix;
    const size_t base = ((size_t)b * C_) * (size_t)(H_ * W_) + (size_t)y * W_ + x;
    const int crow = r + 1;  // center row of the 3-row window
#pragma unroll
    for (int rr = 0; rr < 8; ++rr) {
      int m = rr + half * 8;
      if (m < OUTC) {
        int c = 3 + m;
        float xin = xLds[(c * (ROWS + 2) + crow) * XPITCH + pix + 4];
        out[base + (size_t)c * (H_ * W_)] = xin + acc2[rr];
      }
    }
    if (half == 0) {                     // passthrough channels 0..2
#pragma unroll
      for (int c = 0; c < 3; ++c)
        out[base + (size_t)c * (H_ * W_)] = xLds[(c * (ROWS + 2) + crow) * XPITCH + pix + 4];
    }
    __syncthreads();                     // protect sbuf before next row's Phase B
  }
}

extern "C" void kernel_launch(void* const* d_in, const int* in_sizes, int n_in,
                              void* d_out, int out_size, void* d_ws, size_t ws_size,
                              hipStream_t stream) {
  const float* in = (const float*)d_in[0];
  const float* w1 = (const float*)d_in[1];
  const float* b1 = (const float*)d_in[2];
  const float* w2 = (const float*)d_in[3];
  unsigned short* frags = (unsigned short*)d_ws;   // 10240 bf16 = 20.5 KB

  prep_frags<<<40, 256, 0, stream>>>(w1, b1, w2, frags);
  nca_step<<<B_ * (H_ / ROWS) * (W_ / 128), 256, 0, stream>>>(in, frags, (float*)d_out);
}
